// SelfAttention_36206574306020
// MI455X (gfx1250) — matrix operand order repaired and checked by reading, not compiled, not run on hardware
//
#include <hip/hip_runtime.h>

typedef __attribute__((ext_vector_type(16))) __bf16 v16bf;
typedef __attribute__((ext_vector_type(8)))  __bf16 v8bf;
typedef __attribute__((ext_vector_type(8)))  float  v8f;
typedef int v4i __attribute__((vector_size(16)));   // matches builtin param type

#define B_   4
#define T_   2048
#define KD_  512
#define H_   8
#define NQ_  4096   /* K*H */

#if defined(__HIP_DEVICE_COMPILE__) && defined(__has_builtin)
#if __has_builtin(__builtin_amdgcn_global_load_async_to_lds_b128) && \
    __has_builtin(__builtin_amdgcn_s_wait_asynccnt)
#define ASYNC_LDS 1
#endif
#endif
#ifndef ASYNC_LDS
#define ASYNC_LDS 0
#endif

union BF16x16 { v16bf v; v8bf h[2]; };

__device__ __forceinline__ v8f wmma_bf16(v16bf a, v16bf b, v8f c) {
  // D = A(16x32 bf16) * B(32x16 bf16) + C(16x16 f32)
  return __builtin_amdgcn_wmma_f32_16x16x32_bf16(false, a, false, b, (short)0, c,
                                                 false, false);
}

// A fragment (16x32 bf16): lane L holds row M = L%16.
// elements 0..15 -> K = half*16 + e
__device__ __forceinline__ v16bf load_a_bf16(const __bf16* rowp, int half) {
  BF16x16 u;
  u.h[0] = *(const v8bf*)(rowp + half * 16);
  u.h[1] = *(const v8bf*)(rowp + half * 16 + 8);
  return u.v;
}

// B fragment (32x16 bf16): lane L holds column N = L%16,
// elements = 16 consecutive K starting at half*16 (caller adds half*16).
__device__ __forceinline__ v16bf load_b_bf16(const __bf16* p) {
  BF16x16 u;
  u.h[0] = *(const v8bf*)(p);
  u.h[1] = *(const v8bf*)(p + 8);
  return u.v;
}

// A fragment from an fp32 row, converted to bf16 in registers.
__device__ __forceinline__ v16bf cvt_a_f32(const float* rowp, int half) {
  v16bf r;
#pragma unroll
  for (int e = 0; e < 8; ++e) r[e] = (__bf16)rowp[half * 16 + e];
#pragma unroll
  for (int e = 0; e < 8; ++e) r[8 + e] = (__bf16)rowp[half * 16 + 8 + e];
  return r;
}

__device__ __forceinline__ v8f vzero8() {
  v8f z = {0.f, 0.f, 0.f, 0.f, 0.f, 0.f, 0.f, 0.f};
  return z;
}

#if ASYNC_LDS
// Async 16-byte copy global -> LDS (ASYNCcnt-tracked data path).
__device__ __forceinline__ void async_copy16(const void* g, void* l) {
  __builtin_amdgcn_global_load_async_to_lds_b128(
      (__attribute__((address_space(1))) v4i*)g,
      (__attribute__((address_space(3))) v4i*)l, 0, 0);
}
#endif

// ---------------------------------------------------------------------------
// Kernel 0: weight transpose + fp32->bf16.  dst[c][r] = (bf16)src[r][c].
//   z=0..2 : Wq/Wk/Wv (512 x 4096) -> WT (4096 x 512)
//   z=3    : Wo (4096 x 512)       -> WT (512 x 4096)
// ---------------------------------------------------------------------------
__global__ __launch_bounds__(256)
void weight_tr_kernel(const float* __restrict__ Wq, const float* __restrict__ Wk,
                      const float* __restrict__ Wv, const float* __restrict__ Wo,
                      __bf16* __restrict__ WqT, __bf16* __restrict__ WkT,
                      __bf16* __restrict__ WvT, __bf16* __restrict__ WoT) {
  __shared__ __bf16 tile[32][33];
  const int z = blockIdx.z;
  const float* src = (z == 0) ? Wq : (z == 1) ? Wk : (z == 2) ? Wv : Wo;
  __bf16* dst      = (z == 0) ? WqT : (z == 1) ? WkT : (z == 2) ? WvT : WoT;
  const int R = (z < 3) ? KD_ : NQ_;  // src rows
  const int C = (z < 3) ? NQ_ : KD_;  // src cols
  int r0, c0;
  if (z < 3) { r0 = blockIdx.y * 32; c0 = blockIdx.x * 32; }
  else       { r0 = blockIdx.x * 32; c0 = blockIdx.y * 32; }

  const int tr = threadIdx.x >> 5;   // 0..7
  const int tc = threadIdx.x & 31;   // 0..31
#pragma unroll
  for (int j = 0; j < 32; j += 8)
    tile[tr + j][tc] = (__bf16)src[(size_t)(r0 + tr + j) * C + c0 + tc];
  __syncthreads();
#pragma unroll
  for (int j = 0; j < 32; j += 8)
    dst[(size_t)(c0 + tr + j) * R + r0 + tc] = tile[tc][tr + j];
}

// ---------------------------------------------------------------------------
// Kernel 1: QKV projection.  Y = x @ W (M=8192, N=4096, K=512), z selects W.
// Each wave: 2 M-tiles x 4 N-tiles -> every B fragment feeds two WMMAs.
// B fragments come straight from pre-transposed bf16 weights (L2-resident):
// zero LDS, zero barriers.  Epilogue: column c -> (h = c%8, kd = c/8):
//   z==0: Qh[b,h,t,kd] * 1/sqrt(512); z==1: Kh[b,h,t,kd]; z==2: Vt[b,h,kd,t]
// ---------------------------------------------------------------------------
__global__ __launch_bounds__(256)
void proj_qkv_kernel(const float* __restrict__ x,
                     const __bf16* __restrict__ WqT,
                     const __bf16* __restrict__ WkT,
                     const __bf16* __restrict__ WvT,
                     __bf16* __restrict__ Qh,
                     __bf16* __restrict__ Kh,
                     __bf16* __restrict__ Vt) {
  const int tid  = threadIdx.x;
  const int w    = tid >> 5;
  const int lane = tid & 31;
  const int lr   = lane & 15;
  const int half = lane >> 4;
  const int z    = blockIdx.z;
  const __bf16* WT = (z == 0) ? WqT : (z == 1) ? WkT : WvT;

  const int mbase = blockIdx.x * 256 + w * 32;  // 8 waves x 32 rows
  const int n0    = blockIdx.y * 64;

  const float* xrow0 = x + (size_t)(mbase + lr) * KD_;
  const float* xrow1 = x + (size_t)(mbase + 16 + lr) * KD_;
  const __bf16* wrow[4];
#pragma unroll
  for (int nt = 0; nt < 4; ++nt)
    wrow[nt] = WT + (size_t)(n0 + nt * 16 + lr) * KD_ + half * 16;

  v8f acc[2][4];
#pragma unroll
  for (int mt = 0; mt < 2; ++mt)
#pragma unroll
    for (int nt = 0; nt < 4; ++nt) acc[mt][nt] = vzero8();

#pragma unroll 2
  for (int kk = 0; kk < KD_; kk += 32) {
    v16bf a0 = cvt_a_f32(xrow0 + kk, half);
    v16bf a1 = cvt_a_f32(xrow1 + kk, half);
#pragma unroll
    for (int nt = 0; nt < 4; ++nt) {
      v16bf b = load_b_bf16(wrow[nt] + kk);
      acc[0][nt] = wmma_bf16(a0, b, acc[0][nt]);
      acc[1][nt] = wmma_bf16(a1, b, acc[1][nt]);
    }
  }

  const float qs = 0.044194173824159216f;  // 1/sqrt(512)
#pragma unroll
  for (int mt = 0; mt < 2; ++mt) {
    const int m0 = mbase + mt * 16 + half * 8;  // 8 consecutive rows per lane
    const int b  = m0 >> 11;
    const int t0 = m0 & (T_ - 1);
    if (z == 2) {
#pragma unroll
      for (int nt = 0; nt < 4; ++nt) {
        const int c  = n0 + nt * 16 + lr;
        const int bh = b * H_ + (c & 7);
        const int kd = c >> 3;
        v8bf pk;
#pragma unroll
        for (int r = 0; r < 8; ++r) pk[r] = (__bf16)acc[mt][nt][r];
        *(v8bf*)&Vt[((size_t)bh * KD_ + kd) * T_ + t0] = pk;   // 16B store
      }
    } else if (z == 0) {
#pragma unroll
      for (int nt = 0; nt < 4; ++nt) {
        const int c  = n0 + nt * 16 + lr;
        const int bh = b * H_ + (c & 7);
        const int kd = c >> 3;
#pragma unroll
        for (int r = 0; r < 8; ++r)
          Qh[((size_t)bh * T_ + t0 + r) * KD_ + kd] =
              (__bf16)(acc[mt][nt][r] * qs);
      }
    } else {
#pragma unroll
      for (int nt = 0; nt < 4; ++nt) {
        const int c  = n0 + nt * 16 + lr;
        const int bh = b * H_ + (c & 7);
        const int kd = c >> 3;
#pragma unroll
        for (int r = 0; r < 8; ++r)
          Kh[((size_t)bh * T_ + t0 + r) * KD_ + kd] = (__bf16)acc[mt][nt][r];
      }
    }
  }
}

// ---------------------------------------------------------------------------
// Kernel 2: flash attention per (b,h).
// 8 waves = 2 groups of 4, each group owns 16 query rows.
// Per 32-key block: all waves kick off an async V-tile copy to LDS
// (global_load_async_to_lds_b128, ASYNCcnt), then wave sub==0 computes
// S = Q K^T (32 chained WMMAs) + online softmax while the copy is in flight;
// after s_wait_asynccnt + barrier all 4 waves do P @ V from LDS.
// ---------------------------------------------------------------------------
__global__ __launch_bounds__(256)
void flash_attn_kernel(const __bf16* __restrict__ Qh,
                       const __bf16* __restrict__ Kh,
                       const __bf16* __restrict__ Vt,
                       __bf16* __restrict__ Oc) {
  __shared__ __bf16 Vl[KD_][32];    // V tile: [d][key], 32 KB
  __shared__ __bf16 Pl[2][16][32];  // probabilities per group
  __shared__ float  cf[2][16];      // per-row rescale factors
  __shared__ float  li[2][16];      // per-row 1/l

  const int tid   = threadIdx.x;
  const int w     = tid >> 5;
  const int lane  = tid & 31;
  const int lr    = lane & 15;
  const int half  = lane >> 4;
  const int group = w >> 2;
  const int sub   = w & 3;

  const int bh = blockIdx.y;
  const int q0 = blockIdx.x * 32 + group * 16;

  const __bf16* Qb = Qh + (size_t)bh * T_ * KD_;
  const __bf16* Kb = Kh + (size_t)bh * T_ * KD_;
  const __bf16* Vb = Vt + (size_t)bh * KD_ * T_;

  const int dbase = sub * 128;  // this wave's 128-dim output slice

  v8f acc[8];
#pragma unroll
  for (int i = 0; i < 8; ++i) acc[i] = vzero8();

  float mrow[8], lrow[8];
#pragma unroll
  for (int r = 0; r < 8; ++r) { mrow[r] = -3.0e38f; lrow[r] = 0.f; }

  for (int kb = 0; kb < T_; kb += 32) {
    // ---- stage V tile [512 d x 32 keys] into LDS (async, all 8 waves) ----
#pragma unroll
    for (int j = 0; j < 8; ++j) {
      const int c   = tid * 8 + j;  // 2048 chunks of 16B
      const int row = c >> 2;
      const int seg = c & 3;
      const __bf16* g = Vb + (size_t)row * T_ + kb + seg * 8;
      __bf16* l = &Vl[row][seg * 8];
#if ASYNC_LDS
      async_copy16((const void*)g, (void*)l);
#else
      *(v8bf*)l = *(const v8bf*)g;
#endif
    }

    if (sub == 0) {
      // prefetch next key block of K while this one computes
      if (kb + 32 < T_) {
        __builtin_prefetch(Kb + (size_t)(kb + 32 + lr) * KD_, 0, 3);
        __builtin_prefetch(Kb + (size_t)(kb + 48 + lr) * KD_, 0, 3);
      }
      // ---- S = Q(16x512) @ K(32x512)^T ----
      v8f S0 = vzero8(), S1 = vzero8();
#pragma unroll 4
      for (int kk = 0; kk < KD_; kk += 32) {
        v16bf a  = load_a_bf16(Qb + (size_t)(q0 + lr) * KD_ + kk, half);
        v16bf b0 = load_b_bf16(Kb + (size_t)(kb + lr) * KD_ + kk + half * 16);
        v16bf b1 = load_b_bf16(Kb + (size_t)(kb + 16 + lr) * KD_ + kk + half * 16);
        S0 = wmma_bf16(a, b0, S0);
        S1 = wmma_bf16(a, b1, S1);
      }
      // ---- online softmax over the 32 new keys ----
      float mx[8];
#pragma unroll
      for (int r = 0; r < 8; ++r) mx[r] = fmaxf(S0[r], S1[r]);
#pragma unroll
      for (int msk = 1; msk <= 8; msk <<= 1) {
#pragma unroll
        for (int r = 0; r < 8; ++r)
          mx[r] = fmaxf(mx[r], __shfl_xor(mx[r], msk, 32));
      }
      float p0[8], p1[8], c[8], sum[8];
#pragma unroll
      for (int r = 0; r < 8; ++r) {
        const float mn = fmaxf(mrow[r], mx[r]);
        c[r]   = __expf(mrow[r] - mn);
        mrow[r] = mn;
        p0[r]  = __expf(S0[r] - mn);
        p1[r]  = __expf(S1[r] - mn);
        sum[r] = p0[r] + p1[r];
      }
#pragma unroll
      for (int msk = 1; msk <= 8; msk <<= 1) {
#pragma unroll
        for (int r = 0; r < 8; ++r) sum[r] += __shfl_xor(sum[r], msk, 32);
      }
#pragma unroll
      for (int r = 0; r < 8; ++r) lrow[r] = lrow[r] * c[r] + sum[r];
#pragma unroll
      for (int r = 0; r < 8; ++r) {
        const int row = r + half * 8;
        Pl[group][row][lr]      = (__bf16)p0[r];
        Pl[group][row][16 + lr] = (__bf16)p1[r];
      }
      if (lr < 8) cf[group][lr + half * 8] = c[lr];
    }

#if ASYNC_LDS
    __builtin_amdgcn_s_wait_asynccnt(0);
#endif
    __syncthreads();  // V tile + P + cf visible to everyone

    // ---- O = diag(c)*O + P @ V  (each wave: 8 d-tiles = 128 dims) ----
    float cl[8];
#pragma unroll
    for (int r = 0; r < 8; ++r) cl[r] = cf[group][r + half * 8];
    v16bf pa = load_a_bf16(&Pl[group][lr][0], half);
#pragma unroll
    for (int j = 0; j < 8; ++j) {
      v16bf bv = load_b_bf16(&Vl[dbase + j * 16 + lr][half * 16]);
      v8f t = acc[j];
#pragma unroll
      for (int r = 0; r < 8; ++r) t[r] *= cl[r];
      acc[j] = wmma_bf16(pa, bv, t);
    }
    __syncthreads();  // before Vl/Pl/cf are overwritten
  }

  if (sub == 0 && lr < 8) li[group][lr + half * 8] = 1.0f / lrow[lr];
  __syncthreads();

  float lv[8];
#pragma unroll
  for (int r = 0; r < 8; ++r) lv[r] = li[group][r + half * 8];

  const int b  = bh >> 3;
  const int hh = bh & 7;
#pragma unroll
  for (int j = 0; j < 8; ++j) {
    const int d = dbase + j * 16 + lr;
#pragma unroll
    for (int r = 0; r < 8; ++r) {
      const int t = q0 + r + half * 8;
      Oc[((size_t)(b * T_ + t)) * NQ_ + hh * KD_ + d] =
          (__bf16)(acc[j][r] * lv[r]);
    }
  }
}

// ---------------------------------------------------------------------------
// Kernel 3: out = Oc(8192x4096 bf16) @ Wo(4096x512) + bo, via WoT bf16.
// Each wave: 2 M-tiles x 4 N-tiles; pure global->WMMA loop, no LDS/barriers.
// ---------------------------------------------------------------------------
__global__ __launch_bounds__(256)
void out_proj_kernel(const __bf16* __restrict__ Oc,
                     const __bf16* __restrict__ WoT,
                     const float* __restrict__ bo,
                     float* __restrict__ out) {
  const int tid  = threadIdx.x;
  const int w    = tid >> 5;
  const int lane = tid & 31;
  const int lr   = lane & 15;
  const int half = lane >> 4;

  const int mbase = blockIdx.x * 256 + w * 32;
  const int n0    = blockIdx.y * 64;

  const __bf16* arow0 = Oc + (size_t)(mbase + lr) * NQ_;
  const __bf16* arow1 = Oc + (size_t)(mbase + 16 + lr) * NQ_;
  const __bf16* wrow[4];
#pragma unroll
  for (int nt = 0; nt < 4; ++nt)
    wrow[nt] = WoT + (size_t)(n0 + nt * 16 + lr) * NQ_ + half * 16;

  v8f acc[2][4];
#pragma unroll
  for (int mt = 0; mt < 2; ++mt)
#pragma unroll
    for (int nt = 0; nt < 4; ++nt) acc[mt][nt] = vzero8();

#pragma unroll 2
  for (int kk = 0; kk < NQ_; kk += 32) {
    v16bf a0 = load_a_bf16(arow0 + kk, half);
    v16bf a1 = load_a_bf16(arow1 + kk, half);
#pragma unroll
    for (int nt = 0; nt < 4; ++nt) {
      v16bf b = load_b_bf16(wrow[nt] + kk);
      acc[0][nt] = wmma_bf16(a0, b, acc[0][nt]);
      acc[1][nt] = wmma_bf16(a1, b, acc[1][nt]);
    }
  }

#pragma unroll
  for (int mt = 0; mt < 2; ++mt) {
#pragma unroll
    for (int nt = 0; nt < 4; ++nt) {
      const int c = n0 + nt * 16 + lr;
      const float bias = bo[c];
#pragma unroll
      for (int r = 0; r < 8; ++r) {
        const int m = mbase + mt * 16 + half * 8 + r;
        out[(size_t)m * KD_ + c] = acc[mt][nt][r] + bias;
      }
    }
  }
}

// ---------------------------------------------------------------------------
extern "C" void kernel_launch(void* const* d_in, const int* in_sizes, int n_in,
                              void* d_out, int out_size, void* d_ws,
                              size_t ws_size, hipStream_t stream) {
  (void)in_sizes; (void)n_in; (void)out_size; (void)ws_size;
  const float* x  = (const float*)d_in[0];
  const float* Wq = (const float*)d_in[1];
  const float* Wk = (const float*)d_in[2];
  const float* Wv = (const float*)d_in[3];
  const float* Wo = (const float*)d_in[4];
  const float* bo = (const float*)d_in[5];
  float* out = (float*)d_out;

  const size_t TEN = (size_t)B_ * H_ * T_ * KD_;  // 33,554,432 elems
  const size_t WEL = (size_t)KD_ * NQ_;           //  2,097,152 elems
  __bf16* ws  = (__bf16*)d_ws;
  __bf16* Qh  = ws;             // [b,h,t,k], pre-scaled by 1/sqrt(512)
  __bf16* Kh  = ws + TEN;       // [b,h,t,k]
  __bf16* Vt  = ws + 2 * TEN;   // [b,h,k,t] (transposed)
  __bf16* Oc  = ws + 3 * TEN;   // [b,t,h*k]
  __bf16* WqT = ws + 4 * TEN;           // [4096][512] bf16
  __bf16* WkT = WqT + WEL;
  __bf16* WvT = WkT + WEL;
  __bf16* WoT = WvT + WEL;              // [512][4096] bf16

  weight_tr_kernel<<<dim3(128, 16, 4), 256, 0, stream>>>(Wq, Wk, Wv, Wo, WqT,
                                                         WkT, WvT, WoT);
  proj_qkv_kernel<<<dim3(8192 / 256, NQ_ / 64, 3), 256, 0, stream>>>(
      x, WqT, WkT, WvT, Qh, Kh, Vt);
  flash_attn_kernel<<<dim3(T_ / 32, B_ * H_), 256, 0, stream>>>(Qh, Kh, Vt, Oc);
  out_proj_kernel<<<dim3(8192 / 256, KD_ / 64), 256, 0, stream>>>(Oc, WoT, bo,
                                                                  out);
}
